// RicciFlowRegularizer_73005854097894
// MI455X (gfx1250) — compile-verified, matrix operand order checked by reading, not run on hardware
//
#include <hip/hip_runtime.h>
#include <math.h>
#include <float.h>

// RicciFlowRegularizer for MI455X (gfx1250), wave32 + WMMA f16 + async LDS copy.
// N=8192 D=512 K=16. Heavy math via v_wmma_f32_16x16x32_f16; B-matrix staged
// through LDS with global_load_async_to_lds_b128 (ASYNCcnt) and shared by all
// 4 waves of a block to cut L2 traffic 4x (K1 is L2-BW bound otherwise).

typedef __attribute__((ext_vector_type(16))) _Float16 v16h;
typedef __attribute__((ext_vector_type(8)))  _Float16 v8h;
typedef __attribute__((ext_vector_type(8)))  float    v8f;

#define N_PTS 8192
#define DIM   512
#define KNN   16

// CDNA5 16-bit fragment layout (A: M across lanes, K across VGPRs;
// B is the symmetric map, so one loader serves both operands of X@X^T):
//   lanes 0-15 : VGPR0..3 = K 0..7,  VGPR4..7 = K 16..23
//   lanes 16-31: VGPR0..3 = K 8..15, VGPR4..7 = K 24..31
__device__ inline v16h load_frag_g(const _Float16* __restrict__ base, int row,
                                   int kc, int halfsel) {
  const _Float16* p = base + (size_t)row * DIM + kc * 32 + (halfsel ? 8 : 0);
  union { v16h v; v8h h[2]; } u;
  u.h[0] = *(const v8h*)(p);
  u.h[1] = *(const v8h*)(p + 16);
  return u.v;
}

__device__ inline v16h load_frag_lds(const _Float16* p0) {
  union { v16h v; v8h h[2]; } u;
  u.h[0] = *(const v8h*)(p0);        // ds_load_b128
  u.h[1] = *(const v8h*)(p0 + 16);   // ds_load_b128
  return u.v;
}

// ---------------- K0: fp32 -> f16 convert + exact fp32 row norms ----------
__global__ void __launch_bounds__(256)
k0_convert_norm(const float* __restrict__ emb, _Float16* __restrict__ emb16,
                float* __restrict__ sq) {
  const int row = blockIdx.x;
  const int t = threadIdx.x;
  const float* r = emb + (size_t)row * DIM;
  float a = r[t], b = r[t + 256];
  emb16[(size_t)row * DIM + t]       = (_Float16)a;
  emb16[(size_t)row * DIM + t + 256] = (_Float16)b;
  float s = a * a + b * b;
  #pragma unroll
  for (int off = 16; off > 0; off >>= 1) s += __shfl_xor(s, off);
  __shared__ float wsum[8];
  if ((t & 31) == 0) wsum[t >> 5] = s;
  __syncthreads();
  if (t == 0) {
    float tot = 0.f;
    #pragma unroll
    for (int i = 0; i < 8; ++i) tot += wsum[i];
    sq[row] = tot;
  }
}

// ---------------- K1: WMMA distances + async-LDS B staging + top-16 -------
// Block = 64 query rows (4 waves x one 16-row tile each). Per iteration all
// waves share one 16-column B chunk staged in LDS by async copies
// (double-buffered). 2 selection threads per row keep sorted top-16 lists.
#define NIT (N_PTS / 16)   // 512 column chunks

__global__ void __launch_bounds__(128)
k1_knn(const _Float16* __restrict__ emb16, const float* __restrict__ sq,
       int* __restrict__ knn_idx, float* __restrict__ avgd,
       float* __restrict__ part_knn) {
  const int b = blockIdx.x;
  const int rowbase = b * 64;
  const int t = threadIdx.x;          // 0..127
  const int wv = t >> 5;              // wave 0..3 -> row tile
  const int lane = t & 31;
  const int lr = lane & 15;
  const int halfsel = lane >> 4;

  __shared__ _Float16 s_buf[2][16 * DIM];   // 2 x 16KB B-chunk double buffer
  __shared__ float s_dist[64 * 17];         // padded stride 17
  __shared__ float s_md[128 * KNN];
  __shared__ int   s_mi[128 * KNN];
  __shared__ float s_rowsum[64];

  // Persist this wave's A fragments: 16 rows x 512 dims (16 chunks x 8 VGPRs).
  const int arow = rowbase + wv * 16 + lr;
  v16h afr[16];
  #pragma unroll
  for (int kc = 0; kc < 16; ++kc)
    afr[kc] = load_frag_g(emb16, arow, kc, halfsel);

  float sqA[8];
  #pragma unroll
  for (int j = 0; j < 8; ++j)
    sqA[j] = sq[rowbase + wv * 16 + j + (halfsel ? 8 : 0)];

  float bd[KNN]; int bi[KNN];
  #pragma unroll
  for (int j = 0; j < KNN; ++j) { bd[j] = FLT_MAX; bi[j] = 0; }
  const int selrow = t >> 1, selslot = t & 1;

  // Async stage of one 16-column chunk (16 consecutive rows = contiguous
  // 16KB): 128 lanes x 8 b128 copies; INST_OFFSET applies to both addresses.
  const unsigned long long gbase = (unsigned long long)(size_t)emb16 + t * 16ull;
  const unsigned lbase0 = (unsigned)(size_t)(&s_buf[0][0]) + t * 16u;
  const unsigned lbase1 = (unsigned)(size_t)(&s_buf[1][0]) + t * 16u;
#define ASYNC_CP(LA, GA, OFF)                                                 \
  asm volatile("global_load_async_to_lds_b128 %0, %1, off offset:" #OFF      \
               :: "v"(LA), "v"(GA) : "memory")
#define STAGE_CHUNK(LA, CH)                                                   \
  do {                                                                        \
    unsigned long long ga = gbase + (unsigned long long)(CH) * 16384ull;      \
    ASYNC_CP(LA, ga, 0);     ASYNC_CP(LA, ga, 2048);                          \
    ASYNC_CP(LA, ga, 4096);  ASYNC_CP(LA, ga, 6144);                          \
    ASYNC_CP(LA, ga, 8192);  ASYNC_CP(LA, ga, 10240);                         \
    ASYNC_CP(LA, ga, 12288); ASYNC_CP(LA, ga, 14336);                         \
  } while (0)

  STAGE_CHUNK(lbase0, 0);                    // prologue: chunk 0 -> buf 0
  asm volatile("s_wait_asynccnt 0x0" ::: "memory");
  __syncthreads();

  for (int it = 0; it < NIT; ++it) {
    const int p = it & 1;
    if (it + 1 < NIT) {                      // overlap copy of next chunk
      if (p) STAGE_CHUNK(lbase0, it + 1);
      else   STAGE_CHUNK(lbase1, it + 1);
    }

    // 16 WMMAs over K=512 against the shared B chunk in LDS.
    const _Float16* bp = &s_buf[p][(size_t)lr * DIM];
    v8f acc = {};
    #pragma unroll
    for (int kc = 0; kc < 16; ++kc) {
      v16h bfr = load_frag_lds(bp + kc * 32 + (halfsel ? 8 : 0));
      acc = __builtin_amdgcn_wmma_f32_16x16x32_f16(
          false, afr[kc], false, bfr, (short)0, acc, false, false);
    }

    const int cg = it * 16 + lr;             // candidate (column) index
    const float sqB = sq[cg];
    #pragma unroll
    for (int j = 0; j < 8; ++j) {
      const int m = j + (halfsel ? 8 : 0);   // C layout: M=j+8*half, N=lr
      float d2 = sqA[j] + sqB - 2.0f * acc[j];
      float d = (d2 > 0.0f) ? sqrtf(fmaxf(d2, 1e-12f)) : 0.0f;
      if (cg == rowbase + wv * 16 + m) d = FLT_MAX;   // exclude self
      s_dist[(wv * 16 + m) * 17 + lr] = d;
    }
    __syncthreads();

    // 2 threads per row scan 8 columns each; register-only sorted insert.
    #pragma unroll
    for (int i = 0; i < 8; ++i) {
      const int c = selslot * 8 + i;
      const float d = s_dist[selrow * 17 + c];
      if (d < bd[KNN - 1]) {
        bd[KNN - 1] = d; bi[KNN - 1] = it * 16 + c;
        #pragma unroll
        for (int j = KNN - 1; j > 0; --j) {  // one bubble pass re-sorts
          if (bd[j] < bd[j - 1]) {
            float td = bd[j]; bd[j] = bd[j - 1]; bd[j - 1] = td;
            int   ti = bi[j]; bi[j] = bi[j - 1]; bi[j - 1] = ti;
          }
        }
      }
    }
    asm volatile("s_wait_asynccnt 0x0" ::: "memory");  // next buf landed
    __syncthreads();                                   // publish to block
  }
#undef STAGE_CHUNK
#undef ASYNC_CP

  #pragma unroll
  for (int j = 0; j < KNN; ++j) { s_md[t * KNN + j] = bd[j]; s_mi[t * KNN + j] = bi[j]; }
  __syncthreads();

  if (t < 64) {   // merge the row's 2 partial lists
    float fd[KNN]; int fi[KNN];
    #pragma unroll
    for (int j = 0; j < KNN; ++j) { fd[j] = FLT_MAX; fi[j] = 0; }
    for (int s = 0; s < 2; ++s) {
      const int src = (t * 2 + s) * KNN;
      #pragma unroll
      for (int j = 0; j < KNN; ++j) {
        const float d = s_md[src + j];
        const int  ix = s_mi[src + j];
        if (d < fd[KNN - 1]) {
          fd[KNN - 1] = d; fi[KNN - 1] = ix;
          #pragma unroll
          for (int q = KNN - 1; q > 0; --q) {
            if (fd[q] < fd[q - 1]) {
              float td = fd[q]; fd[q] = fd[q - 1]; fd[q - 1] = td;
              int   ti = fi[q]; fi[q] = fi[q - 1]; fi[q - 1] = ti;
            }
          }
        }
      }
    }
    float sum = 0.0f;
    #pragma unroll
    for (int j = 0; j < KNN; ++j) {
      knn_idx[(size_t)(rowbase + t) * KNN + j] = fi[j];
      sum += fd[j];
    }
    avgd[rowbase + t] = sum * (1.0f / KNN);
    s_rowsum[t] = sum;
  }
  __syncthreads();
  if (t == 0) {
    float tot = 0.f;
    #pragma unroll
    for (int i = 0; i < 64; ++i) tot += s_rowsum[i];
    part_knn[b] = tot;    // fixed-order partial (deterministic)
  }
}

// ---------------- K2: per-point neighbor gram + curvature -----------------
// 1 wave per point. Gram = Nb @ Nb^T: A/B lane maps coincide -> one fragment.
__global__ void __launch_bounds__(256)
k2_curv(const _Float16* __restrict__ emb16, const float* __restrict__ sq,
        const int* __restrict__ knn_idx, const float* __restrict__ avgd,
        const float* __restrict__ ref_curv, float* __restrict__ curv_err) {
  const int wv = threadIdx.x >> 5;
  const int lane = threadIdx.x & 31;
  const int lr = lane & 15;
  const int halfsel = lane >> 4;
  const int i = blockIdx.x * 8 + wv;

  __shared__ float s_nsq[8][16];
  const int myrow = knn_idx[(size_t)i * KNN + lr];
  if (lane < 16) s_nsq[wv][lr] = sq[myrow];
  __syncthreads();

  v8f acc = {};
  #pragma unroll
  for (int kc = 0; kc < 16; ++kc) {
    v16h f = load_frag_g(emb16, myrow, kc, halfsel);
    acc = __builtin_amdgcn_wmma_f32_16x16x32_f16(
        false, f, false, f, (short)0, acc, false, false);
  }
  float lsum = 0.0f;
  const int n = lr;
  #pragma unroll
  for (int j = 0; j < 8; ++j) {
    const int m = j + (halfsel ? 8 : 0);
    float d2 = s_nsq[wv][m] + s_nsq[wv][n] - 2.0f * acc[j];
    float d = (d2 > 0.0f) ? sqrtf(fmaxf(d2, 1e-12f)) : 0.0f;
    if (m < n) lsum += d;              // upper triangle (symmetric)
  }
  #pragma unroll
  for (int off = 16; off > 0; off >>= 1) lsum += __shfl_xor(lsum, off);
  if (lane == 0) {
    const float inter_mean = lsum * (1.0f / 120.0f);   // K*(K-1)/2
    const float curv = inter_mean / (avgd[i] + 1e-8f);
    const float e = curv - ref_curv[i];
    curv_err[i] = e * e;
  }
}

// ---------------- K3: deterministic fixed-order reductions + loss ---------
__global__ void __launch_bounds__(256)
k3_final(const float* __restrict__ curv_err, const float* __restrict__ part_knn,
         const float* __restrict__ ref_dist, float* __restrict__ out) {
  const int t = threadIdx.x;
  __shared__ float red[256];

  float s = 0.f;
  for (int i = t; i < N_PTS; i += 256) s += curv_err[i];
  red[t] = s; __syncthreads();
  for (int off = 128; off > 0; off >>= 1) { if (t < off) red[t] += red[t + off]; __syncthreads(); }
  const float curv_loss = red[0] * (1.0f / N_PTS);
  __syncthreads();

  s = 0.f;
  for (int i = t; i < N_PTS / 64; i += 256) s += part_knn[i];
  red[t] = s; __syncthreads();
  for (int off = 128; off > 0; off >>= 1) { if (t < off) red[t] += red[t + off]; __syncthreads(); }
  const float knn_mean = red[0] * (1.0f / ((float)N_PTS * KNN));
  __syncthreads();

  s = 0.f;
  for (int i = t; i < N_PTS * KNN; i += 256) s += ref_dist[i];
  red[t] = s; __syncthreads();
  for (int off = 128; off > 0; off >>= 1) { if (t < off) red[t] += red[t + off]; __syncthreads(); }
  if (t == 0) {
    const float ref_mean = red[0] * (1.0f / ((float)N_PTS * KNN));
    const float dm = knn_mean - ref_mean;
    out[0] = curv_loss + 0.1f * dm * dm;
  }
}

// ---------------- launch ---------------------------------------------------
extern "C" void kernel_launch(void* const* d_in, const int* in_sizes, int n_in,
                              void* d_out, int out_size, void* d_ws, size_t ws_size,
                              hipStream_t stream) {
  const float* emb      = (const float*)d_in[0];   // 8192*512 f32
  const float* ref_curv = (const float*)d_in[1];   // 8192 f32
  const float* ref_dist = (const float*)d_in[2];   // 8192*16 f32
  float* out = (float*)d_out;

  char* w = (char*)d_ws;
  size_t off = 0;
  auto carve = [&](size_t bytes) -> void* {
    void* p = w + off;
    off = (off + bytes + 255) & ~(size_t)255;
    return p;
  };
  _Float16* emb16   = (_Float16*)carve((size_t)N_PTS * DIM * sizeof(_Float16));
  float*    sq      = (float*)   carve((size_t)N_PTS * sizeof(float));
  int*      knn_idx = (int*)     carve((size_t)N_PTS * KNN * sizeof(int));
  float*    avgd    = (float*)   carve((size_t)N_PTS * sizeof(float));
  float*    partk   = (float*)   carve((size_t)(N_PTS / 64) * sizeof(float));
  float*    cerr    = (float*)   carve((size_t)N_PTS * sizeof(float));
  (void)ws_size; (void)in_sizes; (void)n_in; (void)out_size;

  k0_convert_norm<<<N_PTS,      256, 0, stream>>>(emb, emb16, sq);
  k1_knn         <<<N_PTS / 64, 128, 0, stream>>>(emb16, sq, knn_idx, avgd, partk);
  k2_curv        <<<N_PTS / 8,  256, 0, stream>>>(emb16, sq, knn_idx, avgd, ref_curv, cerr);
  k3_final       <<<1,          256, 0, stream>>>(cerr, partk, ref_dist, out);
}
